// HeteroGNNRiskModel_80307298501441
// MI455X (gfx1250) — matrix operand order repaired and checked by reading, not compiled, not run on hardware
//
#include <hip/hip_runtime.h>
#include <hip/hip_bf16.h>
#include <math.h>

// ---------------- problem constants ----------------
constexpr int kNC = 400000;
constexpr int kNI = 4000;
constexpr int kE  = 4096;
constexpr int kH  = 128;
constexpr int kHEADS = 4;
constexpr int kHD = 32;

// ---------------- types ----------------
typedef _Float16 f16_t;
typedef f16_t v16h __attribute__((ext_vector_type(16)));
typedef f16_t v8h  __attribute__((ext_vector_type(8)));
typedef float v8f  __attribute__((ext_vector_type(8)));
typedef float v4f  __attribute__((ext_vector_type(4)));

static __device__ __forceinline__ v16h make16(v8h a, v8h b) {
  return __builtin_shufflevector(a, b, 0,1,2,3,4,5,6,7,8,9,10,11,12,13,14,15);
}

// A-fragment (16x32 f16) from f16 LDS row-major buffer (pitch in halves).
// ISA 16-bit A layout: lane m (0-15): e0..7 = K(kbase+0..7), e8..15 = K(kbase+16..23);
// lane m+16: +8 on both ranges.
static __device__ __forceinline__ v16h load_a_f16(const f16_t* buf, int pitch, int kbase) {
  const int l = threadIdx.x & 31;
  const int row = l & 15;
  const int lo  = (l >> 4) * 8;
  const f16_t* p = buf + row * pitch + kbase + lo;
  v8h a0 = *(const v8h*)(p);
  v8h a1 = *(const v8h*)(p + 16);
  return make16(a0, a1);
}

// A-fragment from f32 LDS buffer (convert on the fly).
static __device__ __forceinline__ v16h load_a_f32(const float* buf, int pitch, int kbase) {
  const int l = threadIdx.x & 31;
  const int row = l & 15;
  const int lo  = (l >> 4) * 8;
  const float* p = buf + row * pitch + kbase + lo;
  v16h r;
#pragma unroll
  for (int e = 0; e < 8; ++e) { r[e] = (f16_t)p[e]; r[e + 8] = (f16_t)p[e + 16]; }
  return r;
}

// B-fragment (32x16 f16) for C = A @ W^T, W row-major [OUT x K] in f16 (global).
// B[k][n] = W[ntile*16+n][k]; lane n+16h holds K(kbase+8h+0..7), K(kbase+16+8h+0..7) (two 8-runs of W row).
static __device__ __forceinline__ v16h load_b_w(const f16_t* __restrict__ W, int K,
                                                int ntile, int kbase) {
  const int l = threadIdx.x & 31;
  const int j = ntile * 16 + (l & 15);
  const int khalf = (l >> 4) * 8;
  const f16_t* p = W + (size_t)j * K + kbase + khalf;
  v8h b0 = *(const v8h*)(p);
  v8h b1 = *(const v8h*)(p + 16);
  return make16(b0, b1);
}

static __device__ __forceinline__ v8f wmma_f16(v16h a, v16h b, v8f c) {
  return __builtin_amdgcn_wmma_f32_16x16x32_f16(false, a, false, b, (short)0, c, false, false);
}

// N-outer GEMM: hoists all A-fragments when K<=128 (<=4 frags).
template <int K, int OUT, bool A_IS_F32, typename Epi>
static __device__ __forceinline__ void wave_gemm(const void* bufA, int pitchA,
                                                 const f16_t* __restrict__ W, Epi&& epi) {
  constexpr int KS = K / 32;
  static_assert(KS <= 4, "use wave_gemm_kblk for large K");
  v16h afr[KS];
#pragma unroll
  for (int ks = 0; ks < KS; ++ks) {
    if constexpr (A_IS_F32) afr[ks] = load_a_f32((const float*)bufA, pitchA, ks * 32);
    else                    afr[ks] = load_a_f16((const f16_t*)bufA, pitchA, ks * 32);
  }
  for (int nt = 0; nt < OUT / 16; ++nt) {
    v8f acc = {0.f, 0.f, 0.f, 0.f, 0.f, 0.f, 0.f, 0.f};
#pragma unroll
    for (int ks = 0; ks < KS; ++ks) {
      v16h b = load_b_w(W, K, nt, ks * 32);
      acc = wmma_f16(afr[ks], b, acc);
    }
    epi(nt, acc);
  }
}

// K-outer register-blocked GEMM: holds OUT/16 accumulators, loads each A-frag once.
// Used when K is large (F2: K=512, OUT=128 -> 8 accs, 16 A-frag loads total).
template <int K, int OUT, typename Epi>
static __device__ __forceinline__ void wave_gemm_kblk(const f16_t* bufA, int pitchA,
                                                      const f16_t* __restrict__ W,
                                                      Epi&& epi) {
  constexpr int NT = OUT / 16;
  v8f acc[NT];
#pragma unroll
  for (int nt = 0; nt < NT; ++nt) acc[nt] = (v8f){0.f,0.f,0.f,0.f,0.f,0.f,0.f,0.f};
  for (int ks = 0; ks < K / 32; ++ks) {
    v16h a = load_a_f16(bufA, pitchA, ks * 32);
#pragma unroll
    for (int nt = 0; nt < NT; ++nt) {
      v16h b = load_b_w(W, K, nt, ks * 32);
      acc[nt] = wmma_f16(a, b, acc[nt]);
    }
  }
#pragma unroll
  for (int nt = 0; nt < NT; ++nt) epi(nt, acc[nt]);
}

// ---------------- fused feature-attention pipeline ----------------
// Per wave: 16 rows.  x[IN] -> xp[PD] -> v -> o -> LN -> gelu(FFN 4*PD) -> H -> proj(H) = out
constexpr int kPCF = 132;                 // f32 buffer pitch (floats)
constexpr int kPA  = 520;                 // big f16 buffer pitch (halves), holds up to 512 cols
constexpr int kPB  = 136;                 // small f16 buffer pitch (halves)
constexpr int kWAVE_LDS = 16 * kPCF * 4 * 2 + 16 * kPA * 2 + 16 * kPB * 2;  // 37888 B

template <int IN, int PD>
__global__ __launch_bounds__(128, 1)
void feat_kernel(const float* __restrict__ x, int ntiles,
                 const f16_t* __restrict__ pw,  const float* __restrict__ pb,
                 const f16_t* __restrict__ vw,  const float* __restrict__ vb,
                 const f16_t* __restrict__ ow,  const float* __restrict__ ob,
                 const float* __restrict__ lg,  const float* __restrict__ lb,
                 const f16_t* __restrict__ f1w, const float* __restrict__ f1b,
                 const f16_t* __restrict__ f2w, const float* __restrict__ f2b,
                 const f16_t* __restrict__ fpw, const float* __restrict__ fpb,
                 float* __restrict__ out) {
  constexpr int HID = 4 * PD;
  extern __shared__ __align__(16) char smem[];
  const int wave = threadIdx.x >> 5;
  const int lane = threadIdx.x & 31;
  char* wb = smem + wave * kWAVE_LDS;
  float* bufC  = (float*)(wb);
  float* bufXP = (float*)(wb + 16 * kPCF * 4);
  f16_t* bufA  = (f16_t*)(wb + 2 * 16 * kPCF * 4);
  f16_t* bufB  = (f16_t*)(wb + 2 * 16 * kPCF * 4 + 16 * kPA * 2);

  int tile = blockIdx.x * 4 + wave;
  const bool active = tile < ntiles;
  if (!active) tile = ntiles - 1;          // clamp so all waves run (uniform barriers)
  const int rowBase = tile * 16;

  // stage in x tile [16 x IN] f32, coalesced
  for (int idx = lane; idx < 16 * (IN / 4); idx += 32) {
    int r = idx / (IN / 4), c = (idx % (IN / 4)) * 4;
    *(v4f*)&bufC[r * kPCF + c] = *(const v4f*)&x[(size_t)(rowBase + r) * IN + c];
  }
  __syncthreads();

  // P: xp = x @ pw^T + pb  -> bufXP (f32, kept for residual)
  wave_gemm<IN, PD, true>(bufC, kPCF, pw, [&](int nt, v8f acc) {
    int col = nt * 16 + (lane & 15), lo = (lane >> 4) * 8;
    float b = pb[col];
#pragma unroll
    for (int i = 0; i < 8; ++i) bufXP[(lo + i) * kPCF + col] = acc[i] + b;
  });
  __syncthreads();

  // V: t = xp @ vw^T + vb -> bufB (f16)
  wave_gemm<PD, PD, true>(bufXP, kPCF, vw, [&](int nt, v8f acc) {
    int col = nt * 16 + (lane & 15), lo = (lane >> 4) * 8;
    float b = vb[col];
#pragma unroll
    for (int i = 0; i < 8; ++i) bufB[(lo + i) * kPB + col] = (f16_t)(acc[i] + b);
  });
  __syncthreads();

  // O: attn = t @ ow^T + ob -> bufC (f32)
  wave_gemm<PD, PD, false>(bufB, kPB, ow, [&](int nt, v8f acc) {
    int col = nt * 16 + (lane & 15), lo = (lane >> 4) * 8;
    float b = ob[col];
#pragma unroll
    for (int i = 0; i < 8; ++i) bufC[(lo + i) * kPCF + col] = acc[i] + b;
  });
  __syncthreads();

  // LN(xp + attn) -> bufB (f16).  2 lanes per row.
  {
    const int r = lane >> 1, hh = lane & 1;
    constexpr int HWD = PD / 2;
    float s = 0.f, s2 = 0.f;
    for (int c = 0; c < HWD; ++c) {
      int cc = hh * HWD + c;
      float v = bufXP[r * kPCF + cc] + bufC[r * kPCF + cc];
      s += v; s2 += v * v;
    }
    s  += __shfl_xor(s, 1, 32);
    s2 += __shfl_xor(s2, 1, 32);
    float mean = s / (float)PD;
    float var  = s2 / (float)PD - mean * mean;
    float rstd = rsqrtf(var + 1e-5f);
    for (int c = 0; c < HWD; ++c) {
      int cc = hh * HWD + c;
      float v = bufXP[r * kPCF + cc] + bufC[r * kPCF + cc];
      bufB[r * kPB + cc] = (f16_t)((v - mean) * rstd * lg[cc] + lb[cc]);
    }
  }
  __syncthreads();

  // F1: hid = gelu(h @ f1w^T + f1b) -> bufA (f16)
  wave_gemm<PD, HID, false>(bufB, kPB, f1w, [&](int nt, v8f acc) {
    int col = nt * 16 + (lane & 15), lo = (lane >> 4) * 8;
    float b = f1b[col];
#pragma unroll
    for (int i = 0; i < 8; ++i) {
      float v = acc[i] + b;
      v = 0.5f * v * (1.f + erff(v * 0.70710678118654752f));
      bufA[(lo + i) * kPA + col] = (f16_t)v;
    }
  });
  __syncthreads();

  // F2: o = hid @ f2w^T + f2b -> bufB (f16), width kH.  K-outer register blocking.
  wave_gemm_kblk<HID, kH>(bufA, kPA, f2w, [&](int nt, v8f acc) {
    int col = nt * 16 + (lane & 15), lo = (lane >> 4) * 8;
    float b = f2b[col];
#pragma unroll
    for (int i = 0; i < 8; ++i) bufB[(lo + i) * kPB + col] = (f16_t)(acc[i] + b);
  });
  __syncthreads();

  // FP: out = o @ fpw^T + fpb -> bufC (f32)
  wave_gemm<kH, kH, false>(bufB, kPB, fpw, [&](int nt, v8f acc) {
    int col = nt * 16 + (lane & 15), lo = (lane >> 4) * 8;
    float b = fpb[col];
#pragma unroll
    for (int i = 0; i < 8; ++i) bufC[(lo + i) * kPCF + col] = acc[i] + b;
  });
  __syncthreads();

  if (active) {
    for (int idx = lane; idx < 16 * (kH / 4); idx += 32) {
      int r = idx / (kH / 4), c = (idx % (kH / 4)) * 4;
      *(v4f*)&out[(size_t)(rowBase + r) * kH + c] = *(const v4f*)&bufC[r * kPCF + c];
    }
  }
}

// ---------------- edge q/k/v projections (tiny) ----------------
__global__ void edge_qkv_kernel(const float* __restrict__ ch, const float* __restrict__ ih,
                                const int* __restrict__ esrc, const int* __restrict__ edst,
                                const float* __restrict__ qw, const float* __restrict__ qb,
                                const float* __restrict__ kw, const float* __restrict__ kb,
                                const float* __restrict__ vw, const float* __restrict__ vb,
                                f16_t* __restrict__ q16, f16_t* __restrict__ k16,
                                f16_t* __restrict__ vt16) {
  int idx = blockIdx.x * blockDim.x + threadIdx.x;
  if (idx >= kE * kH) return;
  int e = idx / kH, j = idx % kH;
  const float* cr = ch + (size_t)esrc[e] * kH;
  const float* ir = ih + (size_t)edst[e] * kH;
  float qa = qb[j], ka = kb[j], va = vb[j];
  for (int k = 0; k < kH; ++k) {
    float c = cr[k], iv = ir[k];
    qa += c  * qw[j * kH + k];
    ka += iv * kw[j * kH + k];
    va += iv * vw[j * kH + k];
  }
  q16[(size_t)e * kH + j] = (f16_t)qa;
  k16[(size_t)e * kH + j] = (f16_t)ka;
  int hd = j >> 5, d = j & 31;
  vt16[((size_t)hd * kHD + d) * kE + e] = (f16_t)va;   // [head][dim][edge]
}

// ---------------- flash-attention over edges ----------------
__global__ __launch_bounds__(128, 1)
void edge_attn_kernel(const f16_t* __restrict__ q16, const f16_t* __restrict__ k16,
                      const f16_t* __restrict__ vt16, float* __restrict__ attn) {
  constexpr int PP = 40;
  __shared__ f16_t bufP[4][16 * PP];
  const int wave = threadIdx.x >> 5, lane = threadIdx.x & 31;
  const int wg = blockIdx.x * 4 + wave;
  const int hd = wg >> 8;          // / (kE/16)
  const int qt = wg & 255;
  const int qbase = qt * 16;
  const int dbase = hd * kHD;

  v16h aQ;
  {
    int r = lane & 15, lo = (lane >> 4) * 8;
    const f16_t* p = q16 + (size_t)(qbase + r) * kH + dbase + lo;
    aQ = make16(*(const v8h*)p, *(const v8h*)(p + 16));
  }
  v8f o0 = {0.f,0.f,0.f,0.f,0.f,0.f,0.f,0.f};
  v8f o1 = {0.f,0.f,0.f,0.f,0.f,0.f,0.f,0.f};
  float m[8], l[8];
#pragma unroll
  for (int i = 0; i < 8; ++i) { m[i] = -1e30f; l[i] = 0.f; }
  const float rs = 0.17677669529663687f;   // 1/sqrt(32)
  f16_t* P = bufP[wave];

  for (int kt = 0; kt < kE / 32; ++kt) {
    v8f s0, s1;
#pragma unroll
    for (int t = 0; t < 2; ++t) {
      int j = kt * 32 + t * 16 + (lane & 15);
      int khalf = (lane >> 4) * 8;
      const f16_t* p = k16 + (size_t)j * kH + dbase + khalf;
      v16h bK = make16(*(const v8h*)p, *(const v8h*)(p + 16));
      v8f z = {0.f,0.f,0.f,0.f,0.f,0.f,0.f,0.f};
      v8f s = wmma_f16(aQ, bK, z);
      if (t == 0) s0 = s; else s1 = s;
    }
    float cur[8], p0[8], p1[8], rsum[8];
#pragma unroll
    for (int i = 0; i < 8; ++i) {
      s0[i] *= rs; s1[i] *= rs;
      cur[i] = fmaxf(s0[i], s1[i]);
    }
#pragma unroll
    for (int msk = 1; msk < 16; msk <<= 1)
#pragma unroll
      for (int i = 0; i < 8; ++i) cur[i] = fmaxf(cur[i], __shfl_xor(cur[i], msk, 32));
#pragma unroll
    for (int i = 0; i < 8; ++i) {
      float mn = fmaxf(m[i], cur[i]);
      float sc = __expf(m[i] - mn);
      o0[i] *= sc; o1[i] *= sc; l[i] *= sc;
      p0[i] = __expf(s0[i] - mn);
      p1[i] = __expf(s1[i] - mn);
      rsum[i] = p0[i] + p1[i];
      m[i] = mn;
    }
#pragma unroll
    for (int msk = 1; msk < 16; msk <<= 1)
#pragma unroll
      for (int i = 0; i < 8; ++i) rsum[i] += __shfl_xor(rsum[i], msk, 32);
#pragma unroll
    for (int i = 0; i < 8; ++i) l[i] += rsum[i];

    {
      int col = lane & 15, lo = (lane >> 4) * 8;
#pragma unroll
      for (int i = 0; i < 8; ++i) {
        P[(lo + i) * PP + col]      = (f16_t)p0[i];
        P[(lo + i) * PP + 16 + col] = (f16_t)p1[i];
      }
    }
    __syncthreads();
    v16h aP = load_a_f16(P, PP, 0);
#pragma unroll
    for (int nd = 0; nd < 2; ++nd) {
      int khalf = (lane >> 4) * 8;
      const f16_t* p = vt16 + ((size_t)dbase + nd * 16 + (lane & 15)) * kE + kt * 32 + khalf;
      v16h bV = make16(*(const v8h*)p, *(const v8h*)(p + 16));
      if (nd == 0) o0 = wmma_f16(aP, bV, o0);
      else         o1 = wmma_f16(aP, bV, o1);
    }
    __syncthreads();
  }

  {
    int col = lane & 15, lo = (lane >> 4) * 8;
#pragma unroll
    for (int i = 0; i < 8; ++i) {
      float inv = 1.f / l[i];
      attn[(size_t)(qbase + lo + i) * kH + dbase + col]      = o0[i] * inv;
      attn[(size_t)(qbase + lo + i) * kH + dbase + 16 + col] = o1[i] * inv;
    }
  }
}

// ---------------- scatter / finalize ----------------
__global__ void zero_cnt_kernel(float* __restrict__ cnt) {
  int i = blockIdx.x * blockDim.x + threadIdx.x;
  if (i < kNC) cnt[i] = 0.f;
}
__global__ void zero_agg_rows_kernel(const int* __restrict__ esrc, float* __restrict__ agg) {
  int idx = blockIdx.x * blockDim.x + threadIdx.x;
  if (idx >= kE * kH) return;
  agg[(size_t)esrc[idx / kH] * kH + (idx % kH)] = 0.f;
}
__global__ void edge_out_scatter_kernel(const float* __restrict__ attn,
                                        const int* __restrict__ esrc,
                                        const float* __restrict__ ow,
                                        const float* __restrict__ obias,
                                        float* __restrict__ agg, float* __restrict__ cnt) {
  int idx = blockIdx.x * blockDim.x + threadIdx.x;
  if (idx >= kE * kH) return;
  int e = idx / kH, j = idx % kH;
  const float* ar = attn + (size_t)e * kH;
  float a = obias[j];
  for (int k = 0; k < kH; ++k) a += ar[k] * ow[j * kH + k];
  int src = esrc[e];
  atomicAdd(&agg[(size_t)src * kH + j], a);
  if (j == 0) atomicAdd(&cnt[src], 1.0f);
}
__global__ __launch_bounds__(256)
void final_kernel(const float* __restrict__ ch, const float* __restrict__ agg,
                  const float* __restrict__ cnt, const float* __restrict__ lg,
                  const float* __restrict__ lb, float* __restrict__ out) {
  const int wave = threadIdx.x >> 5, lane = threadIdx.x & 31;
  const size_t row = (size_t)blockIdx.x * 8 + wave;
  if (row >= kNC) return;
  float c = cnt[row];
  float inv = (c > 0.f) ? 1.f / (c + 1e-6f) : 0.f;
  float v[4];
  float s = 0.f, s2 = 0.f;
#pragma unroll
  for (int t = 0; t < 4; ++t) {
    int col = lane + t * 32;
    float a = (c > 0.f) ? agg[row * kH + col] * inv : 0.f;
    v[t] = ch[row * kH + col] + a;
    s += v[t]; s2 += v[t] * v[t];
  }
#pragma unroll
  for (int msk = 1; msk < 32; msk <<= 1) {
    s  += __shfl_xor(s,  msk, 32);
    s2 += __shfl_xor(s2, msk, 32);
  }
  float mean = s / (float)kH;
  float var  = s2 / (float)kH - mean * mean;
  float rstd = rsqrtf(var + 1e-5f);
#pragma unroll
  for (int t = 0; t < 4; ++t) {
    int col = lane + t * 32;
    out[row * kH + col] = (v[t] - mean) * rstd * lg[col] + lb[col];
  }
}

__global__ void f32_to_f16_kernel(const float* __restrict__ s, f16_t* __restrict__ d, int n) {
  int i = blockIdx.x * blockDim.x + threadIdx.x;
  if (i < n) d[i] = (f16_t)s[i];
}

// ---------------- host ----------------
enum {
  I_CX = 0, I_IX, I_ESRC, I_EDST,
  I_CFA_PW, I_CFA_PB, I_CFA_VW, I_CFA_VB, I_CFA_OW, I_CFA_OB, I_CFA_LG, I_CFA_LB,
  I_CFA_F1W, I_CFA_F1B, I_CFA_F2W, I_CFA_F2B,
  I_IFA_PW, I_IFA_PB, I_IFA_VW, I_IFA_VB, I_IFA_OW, I_IFA_OB, I_IFA_LG, I_IFA_LB,
  I_IFA_F1W, I_IFA_F1B, I_IFA_F2W, I_IFA_F2B,
  I_CIA_CPW, I_CIA_CPB, I_CIA_IPW, I_CIA_IPB, I_CIA_QW, I_CIA_QB, I_CIA_KW, I_CIA_KB,
  I_CIA_VW, I_CIA_VB, I_CIA_OW, I_CIA_OB, I_CIA_LG, I_CIA_LB
};

extern "C" void kernel_launch(void* const* d_in, const int* in_sizes, int n_in,
                              void* d_out, int out_size, void* d_ws, size_t ws_size,
                              hipStream_t stream) {
  char* ws = (char*)d_ws;
  size_t off = 0;
  auto carve = [&](size_t bytes) -> char* {
    char* p = ws + off;
    off = (off + bytes + 255) & ~(size_t)255;
    return p;
  };
  float* ch   = (float*)carve((size_t)kNC * kH * 4);
  float* agg  = (float*)carve((size_t)kNC * kH * 4);
  float* cnt  = (float*)carve((size_t)kNC * 4);
  float* ih   = (float*)carve((size_t)kNI * kH * 4);
  float* attn = (float*)carve((size_t)kE * kH * 4);
  f16_t* q16  = (f16_t*)carve((size_t)kE * kH * 2);
  f16_t* k16  = (f16_t*)carve((size_t)kE * kH * 2);
  f16_t* vt16 = (f16_t*)carve((size_t)kE * kH * 2);

  struct WCvt { int idx; int n; f16_t* dst; };
  f16_t* cfa_pw16  = (f16_t*)carve(128 * 128 * 2);
  f16_t* cfa_vw16  = (f16_t*)carve(128 * 128 * 2);
  f16_t* cfa_ow16  = (f16_t*)carve(128 * 128 * 2);
  f16_t* cfa_f1w16 = (f16_t*)carve(512 * 128 * 2);
  f16_t* cfa_f2w16 = (f16_t*)carve(128 * 512 * 2);
  f16_t* ifa_pw16  = (f16_t*)carve(64 * 64 * 2);
  f16_t* ifa_vw16  = (f16_t*)carve(64 * 64 * 2);
  f16_t* ifa_ow16  = (f16_t*)carve(64 * 64 * 2);
  f16_t* ifa_f1w16 = (f16_t*)carve(256 * 64 * 2);
  f16_t* ifa_f2w16 = (f16_t*)carve(128 * 256 * 2);
  f16_t* cpw16     = (f16_t*)carve(128 * 128 * 2);
  f16_t* ipw16     = (f16_t*)carve(128 * 128 * 2);

  const WCvt cvts[] = {
    {I_CFA_PW, 128 * 128, cfa_pw16}, {I_CFA_VW, 128 * 128, cfa_vw16},
    {I_CFA_OW, 128 * 128, cfa_ow16}, {I_CFA_F1W, 512 * 128, cfa_f1w16},
    {I_CFA_F2W, 128 * 512, cfa_f2w16},
    {I_IFA_PW, 64 * 64, ifa_pw16}, {I_IFA_VW, 64 * 64, ifa_vw16},
    {I_IFA_OW, 64 * 64, ifa_ow16}, {I_IFA_F1W, 256 * 64, ifa_f1w16},
    {I_IFA_F2W, 128 * 256, ifa_f2w16},
    {I_CIA_CPW, 128 * 128, cpw16}, {I_CIA_IPW, 128 * 128, ipw16},
  };
  for (const auto& c : cvts) {
    f32_to_f16_kernel<<<(c.n + 255) / 256, 256, 0, stream>>>((const float*)d_in[c.idx],
                                                             c.dst, c.n);
  }

  const float* cx   = (const float*)d_in[I_CX];
  const float* ix   = (const float*)d_in[I_IX];
  const int*   esrc = (const int*)d_in[I_ESRC];
  const int*   edst = (const int*)d_in[I_EDST];
  const size_t smem = 4 * (size_t)kWAVE_LDS;

  // company pipeline -> ch
  {
    int ntiles = kNC / 16;                       // 25000
    int blocks = (ntiles + 3) / 4;               // 6250
    feat_kernel<128, 128><<<blocks, 128, smem, stream>>>(
        cx, ntiles,
        cfa_pw16, (const float*)d_in[I_CFA_PB],
        cfa_vw16, (const float*)d_in[I_CFA_VB],
        cfa_ow16, (const float*)d_in[I_CFA_OB],
        (const float*)d_in[I_CFA_LG], (const float*)d_in[I_CFA_LB],
        cfa_f1w16, (const float*)d_in[I_CFA_F1B],
        cfa_f2w16, (const float*)d_in[I_CFA_F2B],
        cpw16, (const float*)d_in[I_CIA_CPB],
        ch);
  }
  // industry pipeline -> ih
  {
    int ntiles = kNI / 16;                       // 250
    int blocks = (ntiles + 3) / 4;               // 63
    feat_kernel<64, 64><<<blocks, 128, smem, stream>>>(
        ix, ntiles,
        ifa_pw16, (const float*)d_in[I_IFA_PB],
        ifa_vw16, (const float*)d_in[I_IFA_VB],
        ifa_ow16, (const float*)d_in[I_IFA_OB],
        (const float*)d_in[I_IFA_LG], (const float*)d_in[I_IFA_LB],
        ifa_f1w16, (const float*)d_in[I_IFA_F1B],
        ifa_f2w16, (const float*)d_in[I_IFA_F2B],
        ipw16, (const float*)d_in[I_CIA_IPB],
        ih);
  }

  edge_qkv_kernel<<<(kE * kH + 255) / 256, 256, 0, stream>>>(
      ch, ih, esrc, edst,
      (const float*)d_in[I_CIA_QW], (const float*)d_in[I_CIA_QB],
      (const float*)d_in[I_CIA_KW], (const float*)d_in[I_CIA_KB],
      (const float*)d_in[I_CIA_VW], (const float*)d_in[I_CIA_VB],
      q16, k16, vt16);

  edge_attn_kernel<<<(kE / 16) * kHEADS / 4, 128, 0, stream>>>(q16, k16, vt16, attn);

  zero_cnt_kernel<<<(kNC + 255) / 256, 256, 0, stream>>>(cnt);
  zero_agg_rows_kernel<<<(kE * kH + 255) / 256, 256, 0, stream>>>(esrc, agg);
  edge_out_scatter_kernel<<<(kE * kH + 255) / 256, 256, 0, stream>>>(
      attn, esrc, (const float*)d_in[I_CIA_OW], (const float*)d_in[I_CIA_OB], agg, cnt);

  final_kernel<<<(kNC + 7) / 8, 256, 0, stream>>>(
      ch, agg, cnt, (const float*)d_in[I_CIA_LG], (const float*)d_in[I_CIA_LB],
      (float*)d_out);

  (void)in_sizes; (void)n_in; (void)out_size; (void)ws_size;
}
